// LSTMSeq2Seq_86182813761972
// MI455X (gfx1250) — matrix operand order repaired and checked by reading, not compiled, not run on hardware
//
#include <hip/hip_runtime.h>

// ---------------- types for WMMA (CDNA5 / gfx1250, wave32) ----------------
typedef __attribute__((ext_vector_type(16))) __bf16 v16bf;
typedef __attribute__((ext_vector_type(8)))  __bf16 v8bf;
typedef __attribute__((ext_vector_type(8)))  float  v8f;

#define BB 16      // batch
#define SS 512     // seq len (enc); dec uses SS-1 steps
#define EE 256     // embed dim
#define HH 512     // hidden
#define GG 2048    // 4*H gates
#define VV 32000   // vocab out

// float -> bf16 (round-to-nearest-even) without relying on hw cvt lowering
__device__ __forceinline__ __bf16 to_bf16(float f) {
  union { float f; unsigned int u; } a; a.f = f;
  unsigned int r = a.u + 0x7FFFu + ((a.u >> 16) & 1u);
  union { unsigned short s; __bf16 b; } o; o.s = (unsigned short)(r >> 16);
  return o.b;
}

__device__ __forceinline__ v16bf cat8(v8bf lo, v8bf hi) {
  return __builtin_shufflevector(lo, hi, 0,1,2,3,4,5,6,7,8,9,10,11,12,13,14,15);
}

// A-matrix 16x32 bf16 fragment (ISA 7.12.2): lane L<16 -> row L, K = k0..k0+7 and k0+16..23
// lane L>=16 -> row L-16, K = k0+8..15 and k0+24..31. A is row-major with leading dim lda.
__device__ __forceinline__ v16bf load_a_frag(const __bf16* A, int lda, int lane) {
  const __bf16* p = A + (lane & 15) * (size_t)lda + ((lane >> 4) << 3);
  v8bf lo = *(const v8bf*)p;
  v8bf hi = *(const v8bf*)(p + 16);
  return cat8(lo, hi);
}

// B-matrix 32x16 bf16 fragment: B = W^T, W row-major (N x K, leading dim ldw).
// lane L: column n = L&15 -> W row n; lanes 0-15 hold K=k0..k0+7 and k0+16..23, lanes 16-31 K=k0+8..15 and k0+24..31.
__device__ __forceinline__ v16bf load_b_frag(const __bf16* W, int ldw, int lane) {
  const __bf16* p = W + (lane & 15) * (size_t)ldw + ((lane >> 4) << 3);
  v8bf lo = *(const v8bf*)p;
  v8bf hi = *(const v8bf*)(p + 16);
  return cat8(lo, hi);
}

__device__ __forceinline__ v8f wmma_bf16(v16bf a, v16bf b, v8f c) {
  // (neg_a, A, neg_b, B, c_mod, C, reuse_a, reuse_b)
  return __builtin_amdgcn_wmma_f32_16x16x32_bf16(false, a, false, b, (short)0, c, false, false);
}

__device__ __forceinline__ float sigmoidf_(float x) { return 1.0f / (1.0f + __expf(-x)); }

// ---------------- elementwise helpers ----------------
__global__ void f32_to_bf16_kernel(const float* __restrict__ in, __bf16* __restrict__ out, int n) {
  int i = blockIdx.x * blockDim.x + threadIdx.x;
  if (i < n) out[i] = to_bf16(in[i]);
}

// X[m, e] = bf16(emb[ids[b*rpi + t], e]),  m = b*rpo + t
__global__ void gather_embed_kernel(const int* __restrict__ ids, const float* __restrict__ emb,
                                    __bf16* __restrict__ X, int nrows, int rpo, int rpi) {
  int i = blockIdx.x * blockDim.x + threadIdx.x;
  if (i >= nrows * EE) return;
  int m = i >> 8;           // / EE (EE==256)
  int e = i & 255;
  int b = m / rpo, t = m - b * rpo;
  int id = ids[b * rpi + t];
  X[(size_t)m * EE + e] = to_bf16(emb[(size_t)id * EE + e]);
}

__global__ void zero_t0_kernel(float* __restrict__ out) {
  int i = blockIdx.x * blockDim.x + threadIdx.x;
  if (i < BB * VV) {
    int b = i / VV, v = i - b * VV;
    out[(size_t)b * SS * VV + v] = 0.0f;
  }
}

// ---------------- xs = X (M x 256) * Wih^T (256 x 2048) + (bih + bhh) ----------------
__global__ void gemm_xs_kernel(const __bf16* __restrict__ X, const __bf16* __restrict__ W,
                               const float* __restrict__ b0, const float* __restrict__ b1,
                               float* __restrict__ xs, int M) {
  int gw   = (int)((blockIdx.x * blockDim.x + threadIdx.x) >> 5);
  int lane = threadIdx.x & 31;
  const int NT = GG / 16;          // 128 column tiles
  int mt = gw / NT, nt = gw - mt * NT;
  if (mt * 16 >= M) return;
  v8f acc = {};
  const __bf16* A  = X + (size_t)mt * 16 * EE;
  const __bf16* Wp = W + (size_t)nt * 16 * EE;
#pragma unroll
  for (int k0 = 0; k0 < EE; k0 += 32)
    acc = wmma_bf16(load_a_frag(A + k0, EE, lane), load_b_frag(Wp + k0, EE, lane), acc);
  int col = nt * 16 + (lane & 15);
  float bias = b0[col] + b1[col];
  int rbase = (lane >> 4) << 3;
#pragma unroll
  for (int v = 0; v < 8; ++v) {
    int m = mt * 16 + v + rbase;
    xs[(size_t)m * GG + col] = acc[v] + bias;
  }
}

// ---------------- persistent LSTM recurrence (single WGP, 32 waves) ----------------
// h kept in LDS as bf16 (16 x 512); c kept in each wave's v8f registers.
// Wave w owns hidden columns [16w, 16w+16): 4 gate tiles (i,f,g,o) per step.
template<bool STORE>
__device__ __forceinline__ void lstm_phase(const float* __restrict__ xs,
                                           const __bf16* __restrict__ Whh,
                                           __bf16* __restrict__ hs_out,
                                           int steps, int rpb,
                                           __bf16* hbuf, v8f& c, int wave, int lane) {
  const int jn    = wave << 4;          // column-tile base
  const int j     = jn + (lane & 15);   // this lane's hidden column
  const int rbase = (lane >> 4) << 3;   // C-layout row base (0 or 8)
  for (int t = 0; t < steps; ++t) {
    v8f acc[4];
    // gates init = xs[b, t, gate*H + j]  (C-tile layout loads)
#pragma unroll
    for (int g = 0; g < 4; ++g) {
      int col = g * HH + j;
#pragma unroll
      for (int v = 0; v < 8; ++v) {
        int b = v + rbase;
        acc[g][v] = xs[((size_t)b * rpb + t) * GG + col];
      }
    }
    // gates += h @ Whh^T  (A fragment shared across the 4 gate tiles)
    for (int k0 = 0; k0 < HH; k0 += 32) {
      v16bf a = load_a_frag(hbuf + k0, HH, lane);
#pragma unroll
      for (int g = 0; g < 4; ++g) {
        const __bf16* wp = Whh + (size_t)(g * HH + jn) * HH + k0;
        acc[g] = wmma_bf16(a, load_b_frag(wp, HH, lane), acc[g]);
      }
    }
    __syncthreads();                    // everyone done reading h for this step
#pragma unroll
    for (int v = 0; v < 8; ++v) {
      float iv = sigmoidf_(acc[0][v]);
      float fv = sigmoidf_(acc[1][v]);
      float gv = tanhf(acc[2][v]);
      float ov = sigmoidf_(acc[3][v]);
      float cv = fv * c[v] + iv * gv;
      c[v] = cv;
      float hv = ov * tanhf(cv);
      int b = v + rbase;
      __bf16 hb = to_bf16(hv);
      hbuf[b * HH + j] = hb;
      if (STORE) hs_out[((size_t)b * rpb + t) * HH + j] = hb;
    }
    __syncthreads();                    // new h visible before next step
  }
}

__global__ __launch_bounds__(1024) void lstm_seq_kernel(const float* __restrict__ xsE,
                                                        const __bf16* __restrict__ WhhE,
                                                        const float* __restrict__ xsD,
                                                        const __bf16* __restrict__ WhhD,
                                                        __bf16* __restrict__ hs) {
  __shared__ __bf16 hbuf[BB * HH];
  int tid = threadIdx.x;
  for (int i = tid; i < BB * HH; i += 1024) hbuf[i] = to_bf16(0.0f);
  v8f c = {};
  __syncthreads();
  int wave = tid >> 5, lane = tid & 31;
  lstm_phase<false>(xsE, WhhE, nullptr, SS,     SS,     hbuf, c, wave, lane);
  lstm_phase<true >(xsD, WhhD, hs,      SS - 1, SS - 1, hbuf, c, wave, lane);
}

// ---------------- logits = hs (8176 x 512) * fc_W^T (512 x 32000) + fc_b ----------------
// 2x2 register-blocked: each wave computes a 32x32 output block (4 accumulators).
// Each A fragment feeds 2 WMMAs and each B fragment feeds 2 WMMAs -> half the L2 traffic
// of the 1-tile-per-wave version. MT=511 is odd: last row-block guarded wave-uniformly.
__global__ void fc_kernel(const __bf16* __restrict__ hs, const __bf16* __restrict__ W,
                          const float* __restrict__ fb, float* __restrict__ out) {
  int gw   = (int)((blockIdx.x * blockDim.x + threadIdx.x) >> 5);
  int lane = threadIdx.x & 31;
  const int MT  = (BB * (SS - 1)) / 16;  // 511 row tiles
  const int MT2 = (MT + 1) / 2;          // 256 row blocks
  const int NT2 = (VV / 16) / 2;         // 1000 col blocks
  int mtb = gw / NT2, ntb = gw - mtb * NT2;
  if (mtb >= MT2) return;
  int mt0 = mtb * 2, mt1 = mt0 + 1;
  int nt0 = ntb * 2, nt1 = nt0 + 1;
  const bool hasM1 = (mt1 < MT);         // wave-uniform predicate (EXEC stays all-ones)

  v8f c00 = {}, c01 = {}, c10 = {}, c11 = {};
  const __bf16* A0 = hs + (size_t)mt0 * 16 * HH;
  const __bf16* A1 = hs + (size_t)mt1 * 16 * HH;
  const __bf16* W0 = W  + (size_t)nt0 * 16 * HH;
  const __bf16* W1 = W  + (size_t)nt1 * 16 * HH;
  const int prow = (lane & 15) * HH;     // this lane's row within a fragment panel

#pragma unroll
  for (int k0 = 0; k0 < HH; k0 += 32) {
    // prefetch one cache line ahead on each stream (global_prefetch_b8; speculative, OOB-safe)
    __builtin_prefetch((const void*)(W0 + prow + k0 + 64), 0, 0);
    __builtin_prefetch((const void*)(W1 + prow + k0 + 64), 0, 0);
    __builtin_prefetch((const void*)(A0 + prow + k0 + 64), 0, 0);
    v16bf b0 = load_b_frag(W0 + k0, HH, lane);
    v16bf b1 = load_b_frag(W1 + k0, HH, lane);
    v16bf a0 = load_a_frag(A0 + k0, HH, lane);
    c00 = wmma_bf16(a0, b0, c00);
    c01 = wmma_bf16(a0, b1, c01);
    if (hasM1) {
      v16bf a1 = load_a_frag(A1 + k0, HH, lane);
      c10 = wmma_bf16(a1, b0, c10);
      c11 = wmma_bf16(a1, b1, c11);
    }
  }

  const int rbase = (lane >> 4) << 3;
  auto store_tile = [&](const v8f& acc, int mt, int nt) {
    int col = nt * 16 + (lane & 15);
    float bias = fb[col];
#pragma unroll
    for (int v = 0; v < 8; ++v) {
      int m = mt * 16 + v + rbase;          // m = b*511 + t
      int b = m / (SS - 1);
      int t = m - b * (SS - 1);
      out[(size_t)b * SS * VV + (size_t)(t + 1) * VV + col] = acc[v] + bias;
    }
  };
  store_tile(c00, mt0, nt0);
  store_tile(c01, mt0, nt1);
  if (hasM1) {
    store_tile(c10, mt1, nt0);
    store_tile(c11, mt1, nt1);
  }
}

// ---------------- host-side orchestration ----------------
extern "C" void kernel_launch(void* const* d_in, const int* in_sizes, int n_in,
                              void* d_out, int out_size, void* d_ws, size_t ws_size,
                              hipStream_t stream) {
  (void)in_sizes; (void)n_in; (void)out_size; (void)ws_size;
  const int*   src    = (const int*)  d_in[0];
  const int*   tgt    = (const int*)  d_in[1];
  const float* encEmb = (const float*)d_in[2];
  const float* encWih = (const float*)d_in[3];
  const float* encWhh = (const float*)d_in[4];
  const float* encBih = (const float*)d_in[5];
  const float* encBhh = (const float*)d_in[6];
  const float* decEmb = (const float*)d_in[7];
  const float* decWih = (const float*)d_in[8];
  const float* decWhh = (const float*)d_in[9];
  const float* decBih = (const float*)d_in[10];
  const float* decBhh = (const float*)d_in[11];
  const float* fcW    = (const float*)d_in[12];
  const float* fcB    = (const float*)d_in[13];
  float* out = (float*)d_out;

  const int Menc = BB * SS;        // 8192
  const int Mdec = BB * (SS - 1);  // 8176

  char* ws = (char*)d_ws;
  size_t off = 0;
  auto alloc = [&](size_t bytes) -> void* {
    void* p = ws + off;
    off = (off + bytes + 255) & ~(size_t)255;
    return p;
  };
  __bf16* wihE = (__bf16*)alloc((size_t)GG * EE * 2);
  __bf16* whhE = (__bf16*)alloc((size_t)GG * HH * 2);
  __bf16* wihD = (__bf16*)alloc((size_t)GG * EE * 2);
  __bf16* whhD = (__bf16*)alloc((size_t)GG * HH * 2);
  __bf16* fcWb = (__bf16*)alloc((size_t)VV * HH * 2);
  __bf16* Xe   = (__bf16*)alloc((size_t)Menc * EE * 2);
  __bf16* Xd   = (__bf16*)alloc((size_t)Mdec * EE * 2);
  float*  xsE  = (float*) alloc((size_t)Menc * GG * 4);
  float*  xsD  = (float*) alloc((size_t)Mdec * GG * 4);
  __bf16* hsb  = (__bf16*)alloc((size_t)Mdec * HH * 2);

  auto cvt = [&](const float* in, __bf16* o, int n) {
    f32_to_bf16_kernel<<<(n + 255) / 256, 256, 0, stream>>>(in, o, n);
  };
  cvt(encWih, wihE, GG * EE);
  cvt(encWhh, whhE, GG * HH);
  cvt(decWih, wihD, GG * EE);
  cvt(decWhh, whhD, GG * HH);
  cvt(fcW,    fcWb, VV * HH);

  gather_embed_kernel<<<(Menc * EE + 255) / 256, 256, 0, stream>>>(src, encEmb, Xe, Menc, SS, SS);
  gather_embed_kernel<<<(Mdec * EE + 255) / 256, 256, 0, stream>>>(tgt, decEmb, Xd, Mdec, SS - 1, SS);

  {
    int waves = (Menc / 16) * (GG / 16);
    gemm_xs_kernel<<<(waves + 7) / 8, 256, 0, stream>>>(Xe, wihE, encBih, encBhh, xsE, Menc);
  }
  {
    int waves = (Mdec / 16) * (GG / 16);
    gemm_xs_kernel<<<(waves + 7) / 8, 256, 0, stream>>>(Xd, wihD, decBih, decBhh, xsD, Mdec);
  }

  lstm_seq_kernel<<<1, 1024, 0, stream>>>(xsE, whhE, xsD, whhD, hsb);

  zero_t0_kernel<<<(BB * VV + 255) / 256, 256, 0, stream>>>(out);
  {
    const int MT2 = ((BB * (SS - 1)) / 16 + 1) / 2;  // 256
    const int NT2 = (VV / 16) / 2;                   // 1000
    int waves = MT2 * NT2;                           // 256000
    fc_kernel<<<(waves + 7) / 8, 256, 0, stream>>>(hsb, fcWb, fcB, out);
  }
}
